// GeometryEncoding_v2_16999480557588
// MI455X (gfx1250) — matrix operand order repaired and checked by reading, not compiled, not run on hardware
//
#include <hip/hip_runtime.h>

typedef _Float16 v8h  __attribute__((ext_vector_type(8)));
typedef _Float16 v16h __attribute__((ext_vector_type(16)));
typedef float    v8f  __attribute__((ext_vector_type(8)));
typedef float    v4f  __attribute__((ext_vector_type(4)));

#define NPTS     8192
#define NK       131072   /* 8192*16 */
#define NTILES   65536    /* 8 * 8192 */
#define TSTRIDE  72       /* activation transpose row stride (f16) -> conflict-free */
#define WSTRIDE  72       /* 64x64 weight row stride (f16): 36m mod 64 distinct     */
#define W1STRIDE 24       /* first-layer weight row stride (f16): 12m mod 64 distinct */

// wait for LDS stores to land + compiler reordering barrier (intra-wave transpose)
#define LDS_FENCE() asm volatile("s_wait_dscnt 0" ::: "memory")

__device__ __forceinline__ v8f wmma16(v16h a, v16h b, v8f c) {
  // D = A(16x32 f16) x B(32x16 f16) + C(16x16 f32)
  return __builtin_amdgcn_wmma_f32_16x16x32_f16(false, a, false, b, (short)0, c, false, false);
}

// A-fragment from padded 64x64 f16 weights (row stride WSTRIDE).
// lane<16: row=rt*16+m, cols kt*32+[0..7] and kt*32+16+[0..7]; lane>=16 shifted by hl*8.
__device__ __forceinline__ v16h ldsA(const _Float16* W, int rt, int kt, int m, int hl) {
  const _Float16* p = W + (rt * 16 + m) * WSTRIDE + kt * 32 + hl * 8;
  v8h lo = *(const v8h*)p;
  v8h hi = *(const v8h*)(p + 16);
  v16h a;
#pragma unroll
  for (int i = 0; i < 8; i++) { a[i] = lo[i]; a[i + 8] = hi[i]; }
  return a;
}

// A-fragment for first layers: only K=0..3 (or 0..1) nonzero -> hi half is zero,
// array stores 16 cols (stride W1STRIDE).
__device__ __forceinline__ v16h ldsA1(const _Float16* W, int rt, int m, int hl) {
  const _Float16* p = W + (rt * 16 + m) * W1STRIDE + hl * 8;
  v8h lo = *(const v8h*)p;
  v16h a;
#pragma unroll
  for (int i = 0; i < 8; i++) { a[i] = lo[i]; a[i + 8] = (_Float16)0.f; }
  return a;
}

// B-fragment from transposed activation buffer T[pt][ch] (f16, stride TSTRIDE)
__device__ __forceinline__ v16h ldsB(const _Float16* T, int pt, int hl, int chbase) {
  const _Float16* p = T + pt * TSTRIDE + chbase + hl * 8;
  v8h lo = *(const v8h*)p;
  v8h hi = *(const v8h*)(p + 16);
  v16h b;
#pragma unroll
  for (int i = 0; i < 8; i++) { b[i] = lo[i]; b[i + 8] = hi[i]; }
  return b;
}

// First-layer B operand: only K columns 0..3 of A are nonzero, and they are seen
// by lanes 0-15 as elements 0..3; every other element multiplies a zero weight
// column, so fill them with copies of the real (finite) values -> no zero splat,
// no lane masking.
__device__ __forceinline__ v16h makeB4(float x0, float x1, float x2, float x3) {
  v16h b;
  b[0] = (_Float16)x0; b[1] = (_Float16)x1;
  b[2] = (_Float16)x2; b[3] = (_Float16)x3;
#pragma unroll
  for (int i = 4; i < 16; i += 2) { b[i] = b[0]; b[i + 1] = b[1]; }
  return b;
}

// per-row bias in C-layout: VGPR r holds row rt*16 + hl*8 + r (broadcast LDS read)
__device__ __forceinline__ v8f ldsBias(const float* bp, int rt, int hl) {
  const float* p = bp + rt * 16 + hl * 8;
  v4f a = *(const v4f*)p;
  v4f b = *(const v4f*)(p + 4);
  v8f r;
#pragma unroll
  for (int i = 0; i < 4; i++) { r[i] = a[i]; r[i + 4] = b[i]; }
  return r;
}

// f16 pack first (cvt output is canonical), then packed-f16 ReLU
// (v_pk_max_num_f16, no canonicalize), then transposed LDS store.
__device__ __forceinline__ void reluStoreT(_Float16* T, int pt, int hl, int rt, v8f acc) {
  v8h hh;
#pragma unroll
  for (int i = 0; i < 8; i++) hh[i] = (_Float16)acc[i];
#pragma unroll
  for (int i = 0; i < 8; i++) hh[i] = __builtin_fmaxf16(hh[i], (_Float16)0.f);
  *(v8h*)(T + pt * TSTRIDE + rt * 16 + hl * 8) = hh;
}

__global__ __launch_bounds__(128) void geo_enc_kernel(
    const float* __restrict__ neigh, const float* __restrict__ center, const float* __restrict__ dcoef,
    const float* __restrict__ cf_w1, const float* __restrict__ cf_b1, const float* __restrict__ cf_g,
    const float* __restrict__ cf_be, const float* __restrict__ cf_m, const float* __restrict__ cf_v,
    const float* __restrict__ cf_w2, const float* __restrict__ cf_b2,
    const float* __restrict__ pl_w1, const float* __restrict__ pl_b1, const float* __restrict__ pl_g,
    const float* __restrict__ pl_be, const float* __restrict__ pl_m, const float* __restrict__ pl_v,
    const float* __restrict__ pl_w2, const float* __restrict__ pl_b2,
    const float* __restrict__ ge_w1, const float* __restrict__ ge_b1, const float* __restrict__ ge_g,
    const float* __restrict__ ge_be, const float* __restrict__ ge_m, const float* __restrict__ ge_v,
    const float* __restrict__ ge_w2, const float* __restrict__ ge_b2,
    float* __restrict__ out)
{
  __shared__ __align__(16) _Float16 wCf1[64 * W1STRIDE];
  __shared__ __align__(16) _Float16 wPl1[64 * W1STRIDE];
  __shared__ __align__(16) _Float16 wGe1[64 * WSTRIDE];
  __shared__ __align__(16) _Float16 wCf2[64 * WSTRIDE];
  __shared__ __align__(16) _Float16 wPl2[64 * WSTRIDE];
  __shared__ __align__(16) _Float16 wGe2[64 * WSTRIDE];
  __shared__ __align__(16) float sbCf1[64];
  __shared__ __align__(16) float sbPl1[64];
  __shared__ __align__(16) float sbGe1[64];
  __shared__ __align__(16) float bCf2[64];
  __shared__ __align__(16) float bPl2[64];
  __shared__ __align__(16) float bGe2[64];
  __shared__ __align__(16) _Float16 dLT[16 * TSTRIDE];
  __shared__ __align__(16) _Float16 Tbuf[8 * 16 * TSTRIDE];  // 4 waves x 2 tiles

  const int tid = threadIdx.x;

  // ---- one-time block init: fold BN into first-layer weights/biases, f16-ize
  if (tid < 64) {
    const int r = tid;
    float s = cf_g[r] * rsqrtf(cf_v[r] + 1e-5f);
#pragma unroll
    for (int c = 0; c < 16; c++)
      wCf1[r * W1STRIDE + c] = (c < 4) ? (_Float16)(s * cf_w1[r * 4 + c]) : (_Float16)0.f;
    sbCf1[r] = s * cf_b1[r] + (cf_be[r] - cf_m[r] * s);

    float sp = pl_g[r] * rsqrtf(pl_v[r] + 1e-5f);
#pragma unroll
    for (int c = 0; c < 16; c++)
      wPl1[r * W1STRIDE + c] = (c < 2) ? (_Float16)(sp * pl_w1[r * 2 + c]) : (_Float16)0.f;
    sbPl1[r] = sp * pl_b1[r] + (pl_be[r] - pl_m[r] * sp);

    float sg = ge_g[r] * rsqrtf(ge_v[r] + 1e-5f);
#pragma unroll 8
    for (int c = 0; c < 64; c++) {
      wGe1[r * WSTRIDE + c] = (_Float16)(sg * ge_w1[r * 64 + c]);
      wCf2[r * WSTRIDE + c] = (_Float16)cf_w2[r * 64 + c];
      wPl2[r * WSTRIDE + c] = (_Float16)pl_w2[r * 64 + c];
      wGe2[r * WSTRIDE + c] = (_Float16)ge_w2[r * 64 + c];
    }
    sbGe1[r] = sg * ge_b1[r] + (ge_be[r] - ge_m[r] * sg);
    bCf2[r] = cf_b2[r];
    bPl2[r] = pl_b2[r];
    bGe2[r] = ge_b2[r];
  }
  for (int i = tid; i < 1024; i += 128) {          // d[c,k] -> dLT[k][c]
    int c = i >> 4, k = i & 15;
    dLT[k * TSTRIDE + c] = (_Float16)dcoef[c * 16 + k];
  }
  __syncthreads();

  const int lane = tid & 31;
  const int wv   = tid >> 5;    // 0..3
  const int hl   = lane >> 4;
  const int pt   = lane & 15;   // point within tile (== k index)
  _Float16* T0 = Tbuf + (wv * 2 + 0) * 16 * TSTRIDE;
  _Float16* T1 = Tbuf + (wv * 2 + 1) * 16 * TSTRIDE;

  const int gw = blockIdx.x * 4 + wv;
  const int step = gridDim.x * 4 * 2;   // waves * 2 tiles

#pragma unroll 1
  for (int t = gw * 2; t < NTILES; t += step) {
    const int bi = t >> 13;              // batch (pair never crosses: n0 is even)
    const int n0 = t & (NPTS - 1);
    const int n1 = n0 + 1;

    // ---- per-point geometry for both tiles (fast sqrt/rcp: feeds f16 anyway;
    //      degenerate 0-denominator cases still select exact 0 like nan_to_num)
    float dx0, dy0, dz0, dist0, cosa0, cose0;
    float dx1, dy1, dz1, dist1, cosa1, cose1;
    {
      const float* np0 = neigh + ((bi * 3) * NPTS + n0) * 16 + pt;
      dx0 = np0[0]      - center[(bi * 3 + 0) * NPTS + n0];
      dy0 = np0[NK]     - center[(bi * 3 + 1) * NPTS + n0];
      dz0 = np0[2 * NK] - center[(bi * 3 + 2) * NPTS + n0];
      const float* np1 = np0 + 16;
      dx1 = np1[0]      - center[(bi * 3 + 0) * NPTS + n1];
      dy1 = np1[NK]     - center[(bi * 3 + 1) * NPTS + n1];
      dz1 = np1[2 * NK] - center[(bi * 3 + 2) * NPTS + n1];
      float a0 = dx0 * dx0 + dy0 * dy0, a1 = dx1 * dx1 + dy1 * dy1;
      dist0 = __builtin_amdgcn_sqrtf(a0 + dz0 * dz0);
      dist1 = __builtin_amdgcn_sqrtf(a1 + dz1 * dz1);
      float e0 = __builtin_amdgcn_sqrtf(a0);
      float e1 = __builtin_amdgcn_sqrtf(a1);
      cose0 = (dist0 > 0.f) ? (e0 * __builtin_amdgcn_rcpf(dist0)) : 0.f;
      cose1 = (dist1 > 0.f) ? (e1 * __builtin_amdgcn_rcpf(dist1)) : 0.f;
      cosa0 = (e0 > 0.f) ? (dy0 * __builtin_amdgcn_rcpf(e0)) : 0.f;
      cosa1 = (e1 > 0.f) ? (dy1 * __builtin_amdgcn_rcpf(e1)) : 0.f;
    }

    // ---- Phase A: cf layer1 (bias in C operand; shared A-frag, both tiles)
    {
      v16h b0 = makeB4(dx0, dy0, dz0, dist0);
      v16h b1 = makeB4(dx1, dy1, dz1, dist1);
#pragma unroll
      for (int rt = 0; rt < 4; rt++) {
        v16h a = ldsA1(wCf1, rt, pt, hl);
        v8f sb = ldsBias(sbCf1, rt, hl);
        reluStoreT(T0, pt, hl, rt, wmma16(a, b0, sb));
        reluStoreT(T1, pt, hl, rt, wmma16(a, b1, sb));
      }
    }
    LDS_FENCE();

    // ---- Phase B: cf layer2 -> enc (registers, C-layout, f32)
    v8f encA0[4], encA1[4];
    {
      v16h c00 = ldsB(T0, pt, hl, 0), c01 = ldsB(T0, pt, hl, 32);
      v16h c10 = ldsB(T1, pt, hl, 0), c11 = ldsB(T1, pt, hl, 32);
#pragma unroll
      for (int rt = 0; rt < 4; rt++) {
        v16h a0 = ldsA(wCf2, rt, 0, pt, hl);
        v16h a1 = ldsA(wCf2, rt, 1, pt, hl);
        v8f bb = ldsBias(bCf2, rt, hl);
        encA0[rt] = wmma16(a1, c01, wmma16(a0, c00, bb));
        encA1[rt] = wmma16(a1, c11, wmma16(a0, c10, bb));
      }
    }

    // ---- Phase C: pl layer1 (bias in C operand)
    {
      v16h b0 = makeB4(cosa0, cose0, cosa0, cose0);
      v16h b1 = makeB4(cosa1, cose1, cosa1, cose1);
#pragma unroll
      for (int rt = 0; rt < 4; rt++) {
        v16h a = ldsA1(wPl1, rt, pt, hl);
        v8f sb = ldsBias(sbPl1, rt, hl);
        reluStoreT(T0, pt, hl, rt, wmma16(a, b0, sb));
        reluStoreT(T1, pt, hl, rt, wmma16(a, b1, sb));
      }
    }
    LDS_FENCE();

    // ---- Phase D: pl layer2, all = enc + d*polar (packed f16), store transposed
    {
      v16h c00 = ldsB(T0, pt, hl, 0), c01 = ldsB(T0, pt, hl, 32);
      v16h c10 = ldsB(T1, pt, hl, 0), c11 = ldsB(T1, pt, hl, 32);
#pragma unroll
      for (int rt = 0; rt < 4; rt++) {
        v16h a0 = ldsA(wPl2, rt, 0, pt, hl);
        v16h a1 = ldsA(wPl2, rt, 1, pt, hl);
        v8f bb = ldsBias(bPl2, rt, hl);
        v8f p0 = wmma16(a1, c01, wmma16(a0, c00, bb));
        v8f p1 = wmma16(a1, c11, wmma16(a0, c10, bb));
        v8h dv = *(const v8h*)(dLT + pt * TSTRIDE + rt * 16 + hl * 8);
        v8h e0h, e1h, p0h, p1h, h0, h1;
#pragma unroll
        for (int i = 0; i < 8; i++) {
          e0h[i] = (_Float16)encA0[rt][i];  p0h[i] = (_Float16)p0[i];
          e1h[i] = (_Float16)encA1[rt][i];  p1h[i] = (_Float16)p1[i];
        }
#pragma unroll
        for (int i = 0; i < 8; i++) {
          h0[i] = __builtin_fmaf16(dv[i], p0h[i], e0h[i]);  // v_pk_fma_f16
          h1[i] = __builtin_fmaf16(dv[i], p1h[i], e1h[i]);
        }
        *(v8h*)(T0 + pt * TSTRIDE + rt * 16 + hl * 8) = h0;
        *(v8h*)(T1 + pt * TSTRIDE + rt * 16 + hl * 8) = h1;
      }
    }
    LDS_FENCE();

    // ---- Phase E: ge layer1 (folded BN bias in C operand) + ReLU
    {
      v16h c00 = ldsB(T0, pt, hl, 0), c01 = ldsB(T0, pt, hl, 32);
      v16h c10 = ldsB(T1, pt, hl, 0), c11 = ldsB(T1, pt, hl, 32);
#pragma unroll
      for (int rt = 0; rt < 4; rt++) {
        v16h a0 = ldsA(wGe1, rt, 0, pt, hl);
        v16h a1 = ldsA(wGe1, rt, 1, pt, hl);
        v8f sb = ldsBias(sbGe1, rt, hl);
        reluStoreT(T0, pt, hl, rt, wmma16(a1, c01, wmma16(a0, c00, sb)));
        reluStoreT(T1, pt, hl, rt, wmma16(a1, c11, wmma16(a0, c10, sb)));
      }
    }
    LDS_FENCE();

    // ---- Phase F: ge layer2 -> global stores (coalesced over k = lane%16)
    {
      v16h c00 = ldsB(T0, pt, hl, 0), c01 = ldsB(T0, pt, hl, 32);
      v16h c10 = ldsB(T1, pt, hl, 0), c11 = ldsB(T1, pt, hl, 32);
      const int ob0 = bi * (64 * NK) + n0 * 16 + pt;
      const int ob1 = ob0 + 16;
#pragma unroll
      for (int rt = 0; rt < 4; rt++) {
        v16h a0 = ldsA(wGe2, rt, 0, pt, hl);
        v16h a1 = ldsA(wGe2, rt, 1, pt, hl);
        v8f bb = ldsBias(bGe2, rt, hl);
        v8f r0 = wmma16(a1, c01, wmma16(a0, c00, bb));
        v8f r1 = wmma16(a1, c11, wmma16(a0, c10, bb));
        const int rowb = rt * 16 + hl * 8;
#pragma unroll
        for (int i = 0; i < 8; i++) {
          out[ob0 + (rowb + i) * NK] = r0[i];
          out[ob1 + (rowb + i) * NK] = r1[i];
        }
      }
    }
    LDS_FENCE();
  }
}

extern "C" void kernel_launch(void* const* d_in, const int* in_sizes, int n_in,
                              void* d_out, int out_size, void* d_ws, size_t ws_size,
                              hipStream_t stream) {
  (void)in_sizes; (void)n_in; (void)out_size; (void)d_ws; (void)ws_size;
  const float* neigh  = (const float*)d_in[0];
  const float* center = (const float*)d_in[1];
  const float* dcoef  = (const float*)d_in[2];
  const float* cf_w1 = (const float*)d_in[3];
  const float* cf_b1 = (const float*)d_in[4];
  const float* cf_g  = (const float*)d_in[5];
  const float* cf_be = (const float*)d_in[6];
  const float* cf_m  = (const float*)d_in[7];
  const float* cf_v  = (const float*)d_in[8];
  const float* cf_w2 = (const float*)d_in[9];
  const float* cf_b2 = (const float*)d_in[10];
  const float* pl_w1 = (const float*)d_in[11];
  const float* pl_b1 = (const float*)d_in[12];
  const float* pl_g  = (const float*)d_in[13];
  const float* pl_be = (const float*)d_in[14];
  const float* pl_m  = (const float*)d_in[15];
  const float* pl_v  = (const float*)d_in[16];
  const float* pl_w2 = (const float*)d_in[17];
  const float* pl_b2 = (const float*)d_in[18];
  const float* ge_w1 = (const float*)d_in[19];
  const float* ge_b1 = (const float*)d_in[20];
  const float* ge_g  = (const float*)d_in[21];
  const float* ge_be = (const float*)d_in[22];
  const float* ge_m  = (const float*)d_in[23];
  const float* ge_v  = (const float*)d_in[24];
  const float* ge_w2 = (const float*)d_in[25];
  const float* ge_b2 = (const float*)d_in[26];

  dim3 grid(1024), block(128);
  hipLaunchKernelGGL(geo_enc_kernel, grid, block, 0, stream,
                     neigh, center, dcoef,
                     cf_w1, cf_b1, cf_g, cf_be, cf_m, cf_v, cf_w2, cf_b2,
                     pl_w1, pl_b1, pl_g, pl_be, pl_m, pl_v, pl_w2, pl_b2,
                     ge_w1, ge_b1, ge_g, ge_be, ge_m, ge_v, ge_w2, ge_b2,
                     (float*)d_out);
}
